// SimpleKANLinear_88776974009041
// MI455X (gfx1250) — matrix operand order repaired and checked by reading, not compiled, not run on hardware
//
#include <hip/hip_runtime.h>

// ---- problem constants (match reference) ----
#define TOKENS 8192
#define IN_F   1024
#define OUT_F  1024
#define NGRID  8
#define KSPL   (IN_F * NGRID)   // 8192 flattened spline K

// ---- tiling ----
#define BM 128
#define BN 128
#define BK 64
#define LDK 72    // LDS row stride in bf16 elems: 144B, keeps 16B alignment + bank stagger

typedef __attribute__((ext_vector_type(16))) __bf16 v16bf;
typedef __attribute__((ext_vector_type(8)))  __bf16 v8bf;
typedef __attribute__((ext_vector_type(4)))  __bf16 v4bf;
typedef __attribute__((ext_vector_type(8)))  float  v8f;

static __device__ __forceinline__ __bf16 f2bf(float f) {
  unsigned u = __builtin_bit_cast(unsigned, f);
  u += 0x7FFFu + ((u >> 16) & 1u);             // round-to-nearest-even
  unsigned short h = (unsigned short)(u >> 16);
  return __builtin_bit_cast(__bf16, h);
}

__global__ __launch_bounds__(256)
void kan_fused_wmma(const float* __restrict__ x,
                    const float* __restrict__ W,
                    const float* __restrict__ bias,
                    const float* __restrict__ sp,     // spline_weight [OUT_F][IN_F*NGRID]
                    const float* __restrict__ grid,
                    float* __restrict__ out)
{
  __shared__ __bf16 lA[BM * LDK];
  __shared__ __bf16 lB[BN * LDK];

  const int tid   = threadIdx.x;
  const int lane  = tid & 31;
  const int wave  = tid >> 5;        // 0..7
  const int waveM = wave >> 2;       // 0..1  -> 64 output rows each
  const int waveN = wave & 3;        // 0..3  -> 32 output cols each
  const int r16   = lane & 15;
  const int hi    = lane >> 4;

  const int bm = blockIdx.y * BM;
  const int bn = blockIdx.x * BN;

  v8f acc[4][2] = {};                // 4 M-subtiles x 2 N-subtiles of 16x16 f32

  float gv[NGRID];
  #pragma unroll
  for (int i = 0; i < NGRID; ++i) gv[i] = grid[i];
  const float invd = 1.0f / (2.0f / (NGRID - 1) + 1e-5f);

  // One 16x16x32 bf16 WMMA pair over the current BK-wide LDS tiles.
  auto mma_step = [&]() {
    #pragma unroll
    for (int ks = 0; ks < BK; ks += 32) {
      v16bf afrag[4];
      #pragma unroll
      for (int m = 0; m < 4; ++m) {
        // A 16x32 bf16 layout: lane = row (mod 16); lanes16-31 hold K+8;
        // two contiguous 8-half chunks at K=base and K=base+16.
        const __bf16* pa = &lA[(waveM * 64 + m * 16 + r16) * LDK + ks + hi * 8];
        v8bf a0 = *reinterpret_cast<const v8bf*>(pa);
        v8bf a1 = *reinterpret_cast<const v8bf*>(pa + 16);
        afrag[m] = __builtin_shufflevector(a0, a1,
                     0,1,2,3,4,5,6,7,8,9,10,11,12,13,14,15);
      }
      #pragma unroll
      for (int n = 0; n < 2; ++n) {
        // B 32x16 bf16 layout: lane = column (mod 16); lanes16-31 hold K+8;
        // two contiguous 8-half chunks at K=base and K=base+16 (matches A).
        const __bf16* pb = &lB[(waveN * 32 + n * 16 + r16) * LDK + ks + hi * 8];
        v8bf b0 = *reinterpret_cast<const v8bf*>(pb);
        v8bf b1 = *reinterpret_cast<const v8bf*>(pb + 16);
        v16bf bf = __builtin_shufflevector(b0, b1,
                     0,1,2,3,4,5,6,7,8,9,10,11,12,13,14,15);
        #pragma unroll
        for (int m = 0; m < 4; ++m)
          acc[m][n] = __builtin_amdgcn_wmma_f32_16x16x32_bf16(
              /*neg_a=*/false, afrag[m], /*neg_b=*/false, bf,
              /*c_mod=*/(short)0, acc[m][n],
              /*reuse_a=*/false, /*reuse_b=*/false);
      }
    }
  };

  // ---------------- pass 1: base = silu(x) @ W^T ----------------
  for (int kt = 0; kt < IN_F; kt += BK) {
    #pragma unroll
    for (int p = 0; p < 8; ++p) {
      int idx = tid + p * 256;                 // 2048 float4 per 128x64 tile
      int row = idx >> 4;
      int c4  = (idx & 15) << 2;
      float4 v = *reinterpret_cast<const float4*>(
          x + (size_t)(bm + row) * IN_F + kt + c4);
      float s0 = v.x / (1.0f + __expf(-v.x));
      float s1 = v.y / (1.0f + __expf(-v.y));
      float s2 = v.z / (1.0f + __expf(-v.z));
      float s3 = v.w / (1.0f + __expf(-v.w));
      v4bf ha = { f2bf(s0), f2bf(s1), f2bf(s2), f2bf(s3) };
      *reinterpret_cast<v4bf*>(&lA[row * LDK + c4]) = ha;

      float4 w4 = *reinterpret_cast<const float4*>(
          W + (size_t)(bn + row) * IN_F + kt + c4);
      v4bf hb = { f2bf(w4.x), f2bf(w4.y), f2bf(w4.z), f2bf(w4.w) };
      *reinterpret_cast<v4bf*>(&lB[row * LDK + c4]) = hb;
    }
    __syncthreads();
    mma_step();
    __syncthreads();
  }

  // ------- pass 2: spline = basis(x) @ sw^T, K flattened (i,g) -------
  for (int kt = 0; kt < KSPL; kt += BK) {
    const int fbase = kt >> 3;                 // 8 input features per K-tile
    {
      // generate basis A-tile: 128 rows x 8 features x 8 grid = 128x64 bf16
      int row = tid >> 1;
      int f0  = (tid & 1) << 2;
      float4 xv = *reinterpret_cast<const float4*>(
          x + (size_t)(bm + row) * IN_F + fbase + f0);
      float xa[4] = { xv.x, xv.y, xv.z, xv.w };
      #pragma unroll
      for (int j = 0; j < 4; ++j) {
        v8bf bb;
        #pragma unroll
        for (int g = 0; g < NGRID; ++g) {
          float t = (xa[j] - gv[g]) * invd;
          bb[g] = f2bf(__expf(-t * t));
        }
        *reinterpret_cast<v8bf*>(&lA[row * LDK + (f0 + j) * NGRID]) = bb;
      }
    }
    #pragma unroll
    for (int p = 0; p < 8; ++p) {
      int idx = tid + p * 256;
      int row = idx >> 4;
      int c4  = (idx & 15) << 2;
      float4 v = *reinterpret_cast<const float4*>(
          sp + (size_t)(bn + row) * KSPL + kt + c4);
      v4bf hb = { f2bf(v.x), f2bf(v.y), f2bf(v.z), f2bf(v.w) };
      *reinterpret_cast<v4bf*>(&lB[row * LDK + c4]) = hb;
    }
    __syncthreads();
    mma_step();
    __syncthreads();
  }

  // ---------------- epilogue: + bias, store f32 ----------------
  // C layout: VGPR r -> M = hi*8 + r, N = lane&15 (per 16x16 subtile)
  #pragma unroll
  for (int n = 0; n < 2; ++n) {
    int o = bn + waveN * 32 + n * 16 + r16;
    float bv = bias[o];
    #pragma unroll
    for (int m = 0; m < 4; ++m) {
      int trow = bm + waveM * 64 + m * 16 + hi * 8;
      #pragma unroll
      for (int r = 0; r < 8; ++r) {
        out[(size_t)(trow + r) * OUT_F + o] = acc[m][n][r] + bv;
      }
    }
  }
}

extern "C" void kernel_launch(void* const* d_in, const int* in_sizes, int n_in,
                              void* d_out, int out_size, void* d_ws, size_t ws_size,
                              hipStream_t stream) {
  const float* x    = (const float*)d_in[0];
  const float* W    = (const float*)d_in[1];
  const float* b    = (const float*)d_in[2];
  const float* sw   = (const float*)d_in[3];
  const float* grid = (const float*)d_in[4];
  float* out = (float*)d_out;

  dim3 gdim(OUT_F / BN, TOKENS / BM);   // 8 x 64 = 512 workgroups
  kan_fused_wmma<<<gdim, 256, 0, stream>>>(x, W, b, sw, grid, out);
}
